// SelfAttention_82832739270703
// MI455X (gfx1250) — matrix operand order repaired and checked by reading, not compiled, not run on hardware
//
#include <hip/hip_runtime.h>
#include <hip/hip_bf16.h>

// ---------------------------------------------------------------------------
// Fused self-attention for MI455X (gfx1250, wave32, WMMA).
//   B=4, S=256, HID=1024, NH=8, HD=128, E=HID*NH=8192, SEQ=S*NH=2048
// Pipeline: cvt fp32->f16  ->  3x WMMA proj GEMM  ->  flash-attention  ->
//           WMMA FC GEMM (f32 out).
// gfx1250 features used:
//   * v_wmma_f32_16x16x32_f16 everywhere (f32 accumulate)
//   * global_load_async_to_lds_b128 + s_wait_asynccnt, double-buffered K tiles
//   * DPP row_ror all-reduce for softmax row statistics (no ds_bpermute)
//   * global prefetch in the GEMM main loop
// ---------------------------------------------------------------------------

typedef __attribute__((ext_vector_type(16))) _Float16 v16h;
typedef __attribute__((ext_vector_type(8)))  _Float16 v8h;
typedef __attribute__((ext_vector_type(8)))  float    v8f;

union V16 { v16h v; v8h h[2]; };

#define HID_  1024
#define NH_   8
#define HD_   128
#define E_    8192      // HID*NH
#define B_    4
#define S_    256
#define SEQ_  2048      // S*NH (effective sequence after the view() trick)
#define M_    (B_ * S_) // 1024 rows for the projection / FC GEMMs
#define NB_   (SEQ_ / 32)   // 64 key blocks per head
#define INV_SCALE 0.03125f  // 1/sqrt(1024)

// Q/K/V live in the flat (B, S, E) projection layout. The reference's
// view(B,-1,NH,HD) maps head element (b,h,t,d), t in [0,SEQ):
//   idx = (b*S + t/8)*E + (t%8)*HID + h*HD + d
__device__ __forceinline__ size_t qkv_idx(int b, int h, int t, int d) {
    return (size_t)(b * S_ + (t >> 3)) * E_ + (size_t)((t & 7) * HID_ + h * HD_ + d);
}

// ---- DPP 16-lane-row all-reduce helpers (row_ror:k, k in {1,2,4,8}) -------
// Rotation butterfly: after combining with ror 1,2,4,8 every lane of a 16-lane
// DPP row holds the reduction of all 16 lanes. Wave32's two DPP rows match the
// two lane-halves of the WMMA C/D layout exactly.
template <int K>
__device__ __forceinline__ float dpp_ror(float x) {
    return __int_as_float(__builtin_amdgcn_update_dpp(
        0, __float_as_int(x), 0x120 | K, 0xF, 0xF, true));
}
__device__ __forceinline__ float row_allmax16(float v) {
    v = fmaxf(v, dpp_ror<1>(v));
    v = fmaxf(v, dpp_ror<2>(v));
    v = fmaxf(v, dpp_ror<4>(v));
    v = fmaxf(v, dpp_ror<8>(v));
    return v;
}
__device__ __forceinline__ float row_allsum16(float v) {
    v += dpp_ror<1>(v);
    v += dpp_ror<2>(v);
    v += dpp_ror<4>(v);
    v += dpp_ror<8>(v);
    return v;
}

// ---- async memory->LDS copy (16B), tracked by ASYNCcnt --------------------
__device__ __forceinline__ void async_copy_b128(const _Float16* gsrc,
                                                _Float16* lds_dst) {
    // generic LDS address: low 32 bits are the LDS byte offset
    unsigned lds = (unsigned)(size_t)lds_dst;
    unsigned long long ga = (unsigned long long)(size_t)gsrc;
    asm volatile("global_load_async_to_lds_b128 %0, %1, off"
                 :: "v"(lds), "v"(ga) : "memory");
}
__device__ __forceinline__ void wait_async0() {
    asm volatile("s_wait_asynccnt 0x0" ::: "memory");
}

// ---------------------------------------------------------------------------
// fp32 -> f16 elementwise conversion
// ---------------------------------------------------------------------------
__global__ void cvt_f32_f16(const float* __restrict__ src,
                            _Float16* __restrict__ dst, int n) {
    int i = blockIdx.x * blockDim.x + threadIdx.x;
    if (i < n) dst[i] = (_Float16)src[i];
}

// ---------------------------------------------------------------------------
// NT GEMM:  C[M,N] = A[M,K] * Bw[N,K]^T + bias[N]
// A, Bw row-major f16 (K contiguous) -> torch Linear convention.
// One wave computes a 32x64 tile: 2 A fragments x 4 B fragments = 8 WMMAs per
// 32-deep K step. Fragment layouts used here (same k order in A and B):
//   A 16x32 f16 : lane L -> row L&15 ; halves hold K {0..15} / {16..31}
//   B 32x16 f16 : lane L -> col L&15 ; halves hold K {0..15} / {16..31}
//   C/D 16x16 f32: lane L -> col L&15 ; VGPR r -> row r + 8*(L>=16)
// ---------------------------------------------------------------------------
template <typename OutT>
__global__ __launch_bounds__(128)
void gemm_nt_wmma(const _Float16* __restrict__ A,
                  const _Float16* __restrict__ Bw,
                  const float* __restrict__ bias,
                  OutT* __restrict__ C,
                  int M, int N, int K) {
    const int lane = threadIdx.x & 31;
    const int wave = threadIdx.x >> 5;
    const int half = lane >> 4;   // 0: lanes 0-15, 1: lanes 16-31
    const int l15  = lane & 15;

    const int ntiles = N >> 6;                   // 64-wide wave tiles in N
    const int wt = blockIdx.x * 4 + wave;        // global wave-tile id
    const int tm = wt / ntiles;                  // 32-row tile index
    const int tn = wt % ntiles;
    const int nbase = tn * 64;

    const _Float16* __restrict__ arow0 = A + (size_t)(tm * 32 + l15) * K;
    const _Float16* __restrict__ arow1 = A + (size_t)(tm * 32 + 16 + l15) * K;

    v8f acc[2][4] = {};

    for (int k0 = 0; k0 < K; k0 += 32) {
        __builtin_prefetch(arow0 + k0 + 128, 0, 1);
        __builtin_prefetch(arow1 + k0 + 128, 0, 1);

        V16 a0, a1;
        const int ka = k0 + half * 16;
        a0.h[0] = *(const v8h*)(arow0 + ka);
        a0.h[1] = *(const v8h*)(arow0 + ka + 8);
        a1.h[0] = *(const v8h*)(arow1 + ka);
        a1.h[1] = *(const v8h*)(arow1 + ka + 8);

        const int kb = k0 + half * 16;
#pragma unroll
        for (int j = 0; j < 4; ++j) {
            const _Float16* brow = Bw + (size_t)(nbase + j * 16 + l15) * K;
            v16h bf = *(const v16h*)(brow + kb);  // 32B aligned contiguous
            acc[0][j] = __builtin_amdgcn_wmma_f32_16x16x32_f16(
                false, a0.v, false, bf, (short)0, acc[0][j], false, false);
            acc[1][j] = __builtin_amdgcn_wmma_f32_16x16x32_f16(
                false, a1.v, false, bf, (short)0, acc[1][j], false, false);
        }
    }

    // epilogue: bias add + store (C/D layout -> row-major C)
#pragma unroll
    for (int mi = 0; mi < 2; ++mi) {
#pragma unroll
        for (int j = 0; j < 4; ++j) {
            const int col = nbase + j * 16 + l15;
            const float bv = bias[col];
#pragma unroll
            for (int r = 0; r < 8; ++r) {
                const int m = tm * 32 + mi * 16 + r + half * 8;
                C[(size_t)m * N + col] = (OutT)(acc[mi][j][r] + bv);
            }
        }
    }
}

// ---------------------------------------------------------------------------
// Flash attention over SEQ=2048, HD=128, f16 in / f16 out, f32 accumulate.
// Grid: x = query tile (SEQ/64 = 32), y = b*NH + h (32). Block = 128 = 4 waves.
// Each wave owns 16 queries. K/V tiles are DOUBLE-BUFFERED in LDS: at the top
// of iteration kb the block waits on ASYNCcnt + one barrier, then immediately
// launches the async K copies and V loads for kb+1 into the other buffer, so
// staging overlaps the 16 WMMAs + softmax of the current block. V is stored
// transposed (VtT[d][key]) so PV B-fragments are contiguous 32B LDS loads.
// ---------------------------------------------------------------------------
__global__ __launch_bounds__(128)
void attn_flash_wmma(const _Float16* __restrict__ Q,
                     const _Float16* __restrict__ K,
                     const _Float16* __restrict__ V,
                     _Float16* __restrict__ O) {
    __shared__ __align__(128) _Float16 Kt [2][32 * HD_];    // 2 x 8 KB [key][d]
    __shared__ __align__(128) _Float16 VtT[2][HD_ * 32];    // 2 x 8 KB [d][key]
    __shared__ __align__(128) _Float16 Pt [4 * 16 * 32];    // 4 KB per-wave P

    const int lane = threadIdx.x & 31;
    const int wave = threadIdx.x >> 5;
    const int half = lane >> 4;
    const int l15  = lane & 15;

    const int bh = blockIdx.y;
    const int b  = bh >> 3;
    const int h  = bh & 7;
    const int qb = blockIdx.x * 64 + wave * 16;   // first query row of this wave

    // ---- preload the 4 Q A-fragments (16 queries x 128 dims), reused all loop
    V16 aQ[4];
    {
        const int qrow = qb + l15;
#pragma unroll
        for (int c = 0; c < 4; ++c) {
            const int d0 = c * 32 + half * 16;
            aQ[c].h[0] = *(const v8h*)(Q + qkv_idx(b, h, qrow, d0));
            aQ[c].h[1] = *(const v8h*)(Q + qkv_idx(b, h, qrow, d0 + 8));
        }
    }

    float m8[8], l8[8];
    v8f o[8] = {};                 // 16x128 output accumulator (8 col-chunks)
#pragma unroll
    for (int r = 0; r < 8; ++r) { m8[r] = -3.0e38f; l8[r] = 0.0f; }

    _Float16* Pw = Pt + wave * (16 * 32);

    const int row = threadIdx.x >> 2;          // staging: key row 0..31
    const int seg = (threadIdx.x & 3) * 32;    // staging: 32-d segment

    // ---- prologue: stage key-block 0 into buffer 0
    {
        const int krow = row;
        v8h vv[4];
#pragma unroll
        for (int u = 0; u < 4; ++u) {
            async_copy_b128(K + qkv_idx(b, h, krow, seg + u * 8),
                            &Kt[0][row * HD_ + seg + u * 8]);
            vv[u] = *(const v8h*)(V + qkv_idx(b, h, krow, seg + u * 8));
        }
#pragma unroll
        for (int u = 0; u < 4; ++u)
#pragma unroll
            for (int i = 0; i < 8; ++i)
                VtT[0][(seg + u * 8 + i) * 32 + row] = vv[u][i];
    }

    for (int kb = 0; kb < NB_; ++kb) {
        // current tile ready + every wave done with the *other* buffer
        wait_async0();
        __syncthreads();

        const int cur = kb & 1;
        const int nxt = cur ^ 1;
        const bool more = (kb + 1) < NB_;

        // ---- kick off staging for kb+1 into the other buffer (overlaps MMA)
        v8h vnext[4];
        if (more) {
            const int krow = (kb + 1) * 32 + row;
#pragma unroll
            for (int u = 0; u < 4; ++u) {
                async_copy_b128(K + qkv_idx(b, h, krow, seg + u * 8),
                                &Kt[nxt][row * HD_ + seg + u * 8]);
                vnext[u] = *(const v8h*)(V + qkv_idx(b, h, krow, seg + u * 8));
            }
        }

        // ---- scores: S(16x32) = Q(16x128) . K(32x128)^T, two 16x16 tiles
        v8f s[2] = {{}, {}};
#pragma unroll
        for (int j = 0; j < 2; ++j) {
            const int key = j * 16 + l15;          // B-fragment column
#pragma unroll
            for (int c = 0; c < 4; ++c) {
                const int kd = c * 32 + half * 16;
                v16h bK = *(const v16h*)(&Kt[cur][key * HD_ + kd]);
                s[j] = __builtin_amdgcn_wmma_f32_16x16x32_f16(
                    false, aQ[c].v, false, bK, (short)0, s[j], false, false);
            }
            s[j] *= INV_SCALE;
        }

        // ---- online softmax (per row r of this lane's half), DPP reductions
#pragma unroll
        for (int r = 0; r < 8; ++r) {
            const float mx = row_allmax16(fmaxf(s[0][r], s[1][r]));
            const float mn    = fmaxf(m8[r], mx);
            const float alpha = __expf(m8[r] - mn);
            const float p0    = __expf(s[0][r] - mn);
            const float p1    = __expf(s[1][r] - mn);
            const float rs    = row_allsum16(p0 + p1);
            l8[r] = l8[r] * alpha + rs;
            m8[r] = mn;
#pragma unroll
            for (int ch = 0; ch < 8; ++ch) o[ch][r] *= alpha;
            // park P in LDS (row-major 16x32) to re-fragment into A layout
            const int rg = r + half * 8;
            Pw[rg * 32 + l15]      = (_Float16)p0;
            Pw[rg * 32 + 16 + l15] = (_Float16)p1;
        }
        // intra-wave LDS RAW: stores above feed the cross-lane loads below
        asm volatile("s_wait_dscnt 0x0" ::: "memory");

        // ---- P as A-fragment (16x32 f16)
        V16 aP;
        {
            const int kbase = half * 16;
            aP.h[0] = *(const v8h*)(Pw + l15 * 32 + kbase);
            aP.h[1] = *(const v8h*)(Pw + l15 * 32 + kbase + 8);
        }

        // ---- O(16x128) += P(16x32) . V(32x128): contiguous 32B LDS loads
#pragma unroll
        for (int ch = 0; ch < 8; ++ch) {
            v16h bV = *(const v16h*)(&VtT[cur][(ch * 16 + l15) * 32 + half * 16]);
            o[ch] = __builtin_amdgcn_wmma_f32_16x16x32_f16(
                false, aP.v, false, bV, (short)0, o[ch], false, false);
        }

        // ---- finish staging kb+1: transpose-store V into the other buffer
        if (more) {
#pragma unroll
            for (int u = 0; u < 4; ++u)
#pragma unroll
                for (int i = 0; i < 8; ++i)
                    VtT[nxt][(seg + u * 8 + i) * 32 + row] = vnext[u][i];
        }
    }

    // ---- normalize + store O in (B, SEQ, NH, HD) layout (== FC input rows)
#pragma unroll
    for (int r = 0; r < 8; ++r) {
        const float inv = 1.0f / l8[r];
        const int t = qb + r + half * 8;
#pragma unroll
        for (int ch = 0; ch < 8; ++ch) {
            const int d = ch * 16 + l15;
            O[(size_t)(b * SEQ_ + t) * (NH_ * HD_) + h * HD_ + d] =
                (_Float16)(o[ch][r] * inv);
        }
    }
}

// ---------------------------------------------------------------------------
// Host-side launcher
// ---------------------------------------------------------------------------
extern "C" void kernel_launch(void* const* d_in, const int* in_sizes, int n_in,
                              void* d_out, int out_size, void* d_ws, size_t ws_size,
                              hipStream_t stream) {
    const float* q   = (const float*)d_in[0];
    const float* k_  = (const float*)d_in[1];
    const float* v   = (const float*)d_in[2];
    const float* Wq  = (const float*)d_in[3];
    const float* bq  = (const float*)d_in[4];
    const float* Wk  = (const float*)d_in[5];
    const float* bk  = (const float*)d_in[6];
    const float* Wv  = (const float*)d_in[7];
    const float* bv  = (const float*)d_in[8];
    const float* Wfc = (const float*)d_in[9];
    const float* bfc = (const float*)d_in[10];
    float* out = (float*)d_out;

    // --- workspace carve-out (all f16, all sizes multiples of 64B) ---------
    const size_t NX = (size_t)B_ * S_ * HID_;   // 1,048,576 activation elems
    const size_t NW = (size_t)E_ * HID_;        // 8,388,608 weight elems
    const size_t NQ = (size_t)B_ * S_ * E_;     // 8,388,608 projection elems
    _Float16* ws = (_Float16*)d_ws;
    size_t off = 0;
    auto carve = [&](size_t n) { _Float16* p = ws + off; off += n; return p; };
    _Float16* hXq  = carve(NX);
    _Float16* hXk  = carve(NX);
    _Float16* hXv  = carve(NX);
    _Float16* hWq  = carve(NW);
    _Float16* hWk  = carve(NW);
    _Float16* hWv  = carve(NW);
    _Float16* hWfc = carve(NW);
    _Float16* hQ   = carve(NQ);
    _Float16* hK   = carve(NQ);
    _Float16* hV   = carve(NQ);
    _Float16* hO   = carve(NQ);
    // total: ~134 MB of f16 scratch

    // --- 1) fp32 -> f16 conversions ---------------------------------------
    auto cvt = [&](const float* s, _Float16* d, size_t n) {
        cvt_f32_f16<<<(unsigned)((n + 255) / 256), 256, 0, stream>>>(s, d, (int)n);
    };
    cvt(q,   hXq,  NX);
    cvt(k_,  hXk,  NX);
    cvt(v,   hXv,  NX);
    cvt(Wq,  hWq,  NW);
    cvt(Wk,  hWk,  NW);
    cvt(Wv,  hWv,  NW);
    cvt(Wfc, hWfc, NW);

    // --- 2) Q/K/V projections: (1024 x 8192) = (1024 x 1024) x (8192 x 1024)^T
    {
        const int M = M_, N = E_, K = HID_;
        const unsigned blocks = (unsigned)((size_t)(M / 32) * (N / 64) / 4);
        gemm_nt_wmma<_Float16><<<blocks, 128, 0, stream>>>(hXq, hWq, bq, hQ, M, N, K);
        gemm_nt_wmma<_Float16><<<blocks, 128, 0, stream>>>(hXk, hWk, bk, hK, M, N, K);
        gemm_nt_wmma<_Float16><<<blocks, 128, 0, stream>>>(hXv, hWv, bv, hV, M, N, K);
    }

    // --- 3) flash attention: 32 heads x 2048 seq x 128 hd ------------------
    {
        dim3 grid(SEQ_ / 64, B_ * NH_);
        attn_flash_wmma<<<grid, 128, 0, stream>>>(hQ, hK, hV, hO);
    }

    // --- 4) output FC: (1024 x 1024) = (1024 x 8192) x (1024 x 8192)^T -----
    {
        const int M = M_, N = HID_, K = E_;
        const unsigned blocks = (unsigned)((size_t)(M / 32) * (N / 64) / 4);
        gemm_nt_wmma<float><<<blocks, 128, 0, stream>>>(hO, hWfc, bfc, out, M, N, K);
    }
    (void)in_sizes; (void)n_in; (void)out_size; (void)ws_size;
}
